// NNConvNet_36524401885778
// MI455X (gfx1250) — matrix operand order repaired and checked by reading, not compiled, not run on hardware
//
#include <hip/hip_runtime.h>

typedef __attribute__((ext_vector_type(16))) _Float16 v16h;
typedef __attribute__((ext_vector_type(8)))  float    v8f;

#define NN   25000
#define EE   400000
#define FIN  128
#define FE   16
#define CW   32
#define SLOPE 0.01f

// leaky_relu(v) == max(v, SLOPE*v) for 0 < SLOPE < 1  -> v_mul + v_max (no cndmask)
__device__ __forceinline__ float lrelu(float v) { return fmaxf(v, SLOPE * v); }

// ---------------------------------------------------------------------------
// Kernel 1: h = leaky_relu(x @ W_in + b_in)   [25000,128] @ [128,32]
// One wave = one 16-row tile, both 16-col tiles. K=128 -> 4 chained WMMAs.
// ---------------------------------------------------------------------------
__global__ __launch_bounds__(256) void k_input_proj(
    const float* __restrict__ x, const float* __restrict__ W_in,
    const float* __restrict__ b_in, float* __restrict__ h)
{
  __shared__ _Float16 win[CW * FIN];            // col-major [col][k]
  int tid = threadIdx.x;
  for (int idx = tid; idx < CW * FIN; idx += 256) {
    int col = idx & (CW - 1);
    int k   = idx >> 5;                          // idx = k*32 + col (coalesced read)
    win[col * FIN + k] = (_Float16)W_in[idx];
  }
  __syncthreads();

  int lane = tid & 31, wv = tid >> 5;
  int mtile = blockIdx.x * 8 + wv;
  const int mtiles = (NN + 15) / 16;
  if (mtile >= mtiles) return;

  int n  = lane & 15;
  int hi = lane >> 4;                            // 0: lanes 0-15, 1: lanes 16-31
  int mrow = mtile * 16 + n;
  int mc   = mrow < NN ? mrow : NN - 1;

  // A fragments (contiguous k order, same as B): lanes 0-15 K=0..15; lanes 16-31 K=16..31
  v16h A[4];
#pragma unroll
  for (int kk = 0; kk < 4; ++kk) {
    const float4* xp = (const float4*)(x + (size_t)mc * FIN + kk * 32 + hi * 16);
    float4 a0 = xp[0], a1 = xp[1];               // K = base .. base+7
    float4 a2 = xp[2], a3 = xp[3];               // K = base+8 .. base+15
    A[kk][0]  = (_Float16)a0.x; A[kk][1]  = (_Float16)a0.y;
    A[kk][2]  = (_Float16)a0.z; A[kk][3]  = (_Float16)a0.w;
    A[kk][4]  = (_Float16)a1.x; A[kk][5]  = (_Float16)a1.y;
    A[kk][6]  = (_Float16)a1.z; A[kk][7]  = (_Float16)a1.w;
    A[kk][8]  = (_Float16)a2.x; A[kk][9]  = (_Float16)a2.y;
    A[kk][10] = (_Float16)a2.z; A[kk][11] = (_Float16)a2.w;
    A[kk][12] = (_Float16)a3.x; A[kk][13] = (_Float16)a3.y;
    A[kk][14] = (_Float16)a3.z; A[kk][15] = (_Float16)a3.w;
  }

#pragma unroll
  for (int nt = 0; nt < 2; ++nt) {
    int col = nt * 16 + n;
    v8f c = {};
#pragma unroll
    for (int kk = 0; kk < 4; ++kk) {
      // B frag: lanes 0-15 K=kk*32..+15, lanes 16-31 K=kk*32+16..+31 (contiguous col-major)
      v16h B = *(const v16h*)&win[col * FIN + kk * 32 + hi * 16];
      c = __builtin_amdgcn_wmma_f32_16x16x32_f16(false, A[kk], false, B,
                                                 (short)0, c, false, false);
    }
    float bias = b_in[col];
#pragma unroll
    for (int r = 0; r < 8; ++r) {
      int m = mtile * 16 + hi * 8 + r;
      if (m < NN) h[(size_t)m * CW + col] = lrelu(c[r] + bias);
    }
  }
}

// ---------------------------------------------------------------------------
// Kernel 2 (fused NNConv edge path): per 16-edge tile
//   We_tile = leaky(edge_attr_tile @ W_edge + b_edge) via WMMA (never stored),
//   msg[m,o] += We[m, c*32+o] * h[src[m], c]  accumulated in registers,
//   scatter-add into aggr with hardware f32 global atomics.
// Column-tile j covers cols j*16..j*16+15 -> c = j>>1, o = (j&1)*16 + n, so each
// (m,o) accumulator is owned by exactly one lane across the whole sweep.
// ---------------------------------------------------------------------------
__global__ __launch_bounds__(256) void k_edge_fused(
    const float* __restrict__ edge_attr, const int* __restrict__ edge_index,
    const float* __restrict__ W_edge, const float* __restrict__ b_edge,
    const float* __restrict__ h, float* __restrict__ aggr)
{
  __shared__ _Float16 we[FE * 1024 + 16];       // col-major [col][k] + 16-half zero pad
  __shared__ float    be[1024];
  __shared__ float    hsrc[8][16][CW];
  __shared__ int      dsts[8][16];

  int tid = threadIdx.x;
  for (int idx = tid; idx < FE * 1024; idx += 256) {
    int col = idx & 1023;
    int k   = idx >> 10;                         // idx = k*1024 + col (coalesced read)
    we[col * FE + k] = (_Float16)W_edge[idx];
  }
  for (int idx = tid; idx < 1024; idx += 256) be[idx] = b_edge[idx];
  if (tid < 16) we[FE * 1024 + tid] = (_Float16)0.f;

  int lane = tid & 31, wv = tid >> 5;
  int n  = lane & 15;
  int hi = lane >> 4;
  int e0 = (blockIdx.x * 8 + wv) * 16;           // EE/16 tiles, exact

  // stage h[src] (16 edges x 32 ch) and dst indices; lane = (edge n, half hi)
  {
    int e = e0 + n;
    int s = edge_index[e];                       // row 0 = source
    const float4* hp = (const float4*)(h + (size_t)s * CW + hi * 16);
    float4* dp = (float4*)(&hsrc[wv][n][hi * 16]);
    dp[0] = hp[0]; dp[1] = hp[1]; dp[2] = hp[2]; dp[3] = hp[3];
    if (hi == 0) dsts[wv][n] = edge_index[EE + e];   // row 1 = target
  }
  __syncthreads();

  // A fragment: edge_attr tile [16x16], K padded 16->32 with zeros
  v16h A;
  {
    const float4* eap = (const float4*)(edge_attr + (size_t)(e0 + n) * FE);
    float4 a0 = eap[0], a1 = eap[1], a2 = eap[2], a3 = eap[3];
    A[0] = (_Float16)a0.x; A[1] = (_Float16)a0.y;
    A[2] = (_Float16)a0.z; A[3] = (_Float16)a0.w;
    A[4] = (_Float16)a1.x; A[5] = (_Float16)a1.y;
    A[6] = (_Float16)a1.z; A[7] = (_Float16)a1.w;
    A[8] = (_Float16)a2.x; A[9] = (_Float16)a2.y;
    A[10] = (_Float16)a2.z; A[11] = (_Float16)a2.w;
    A[12] = (_Float16)a3.x; A[13] = (_Float16)a3.y;
    A[14] = (_Float16)a3.z; A[15] = (_Float16)a3.w;
#pragma unroll
    for (int i = 0; i < 16; ++i) if (hi) A[i] = (_Float16)0.f;
  }

  float accA[8], accB[8];
#pragma unroll
  for (int r = 0; r < 8; ++r) { accA[r] = 0.f; accB[r] = 0.f; }

  const int zoff = FE * 1024;                    // zero-pad offset (halfs)

  for (int jc = 0; jc < CW; ++jc) {              // jc = input channel c
    float hv[8];
#pragma unroll
    for (int r = 0; r < 8; ++r) hv[r] = hsrc[wv][hi * 8 + r][jc];   // LDS broadcast
#pragma unroll
    for (int hf = 0; hf < 2; ++hf) {
      int tile = jc * 2 + hf;                    // column-tile, cols tile*16..+15
      int boff = hi ? zoff : (tile * 16 + n) * FE;   // lanes 16-31: zeros (K=16..31)
      v16h B = *(const v16h*)&we[boff];
      v8f z = {};
      v8f c = __builtin_amdgcn_wmma_f32_16x16x32_f16(false, A, false, B,
                                                     (short)0, z, false, false);
      float bias = be[tile * 16 + n];
      float* acc = hf ? accB : accA;             // o = hf*16 + n
#pragma unroll
      for (int r = 0; r < 8; ++r)
        acc[r] = fmaf(lrelu(c[r] + bias), hv[r], acc[r]);
    }
  }

  // scatter-add messages into aggr[dst] with hardware f32 atomics
#pragma unroll
  for (int r = 0; r < 8; ++r) {
    int m = hi * 8 + r;
    int d = dsts[wv][m];
    unsafeAtomicAdd(&aggr[(size_t)d * CW + n],      accA[r]);
    unsafeAtomicAdd(&aggr[(size_t)d * CW + 16 + n], accB[r]);
  }
}

// ---------------------------------------------------------------------------
// Kernel 3: h2 = leaky(aggr + h @ W_root + b_conv);  out = h2 @ W_out + b_out
// K=32 exact WMMA, then shfl_xor butterfly over the 16-lane halves for W_out.
// ---------------------------------------------------------------------------
__global__ __launch_bounds__(256) void k_root_out(
    const float* __restrict__ h, const float* __restrict__ aggr,
    const float* __restrict__ W_root, const float* __restrict__ b_conv,
    const float* __restrict__ W_out, const float* __restrict__ b_out,
    float* __restrict__ out)
{
  __shared__ _Float16 wr[CW * CW];               // col-major
  int tid = threadIdx.x;
  for (int idx = tid; idx < CW * CW; idx += 256) {
    int col = idx & 31, k = idx >> 5;
    wr[col * CW + k] = (_Float16)W_root[idx];
  }
  __syncthreads();

  int lane = tid & 31, wv = tid >> 5;
  int mtile = blockIdx.x * 8 + wv;
  const int mtiles = (NN + 15) / 16;
  if (mtile >= mtiles) return;
  int n = lane & 15, hi = lane >> 4;

  int mrow = mtile * 16 + n;
  int mc   = mrow < NN ? mrow : NN - 1;

  v16h A;
  {
    const float4* hp = (const float4*)(h + (size_t)mc * CW + hi * 16);
    float4 a0 = hp[0], a1 = hp[1];               // K = base .. base+7
    float4 a2 = hp[2], a3 = hp[3];               // K = base+8 .. base+15
    A[0]  = (_Float16)a0.x; A[1]  = (_Float16)a0.y;
    A[2]  = (_Float16)a0.z; A[3]  = (_Float16)a0.w;
    A[4]  = (_Float16)a1.x; A[5]  = (_Float16)a1.y;
    A[6]  = (_Float16)a1.z; A[7]  = (_Float16)a1.w;
    A[8]  = (_Float16)a2.x; A[9]  = (_Float16)a2.y;
    A[10] = (_Float16)a2.z; A[11] = (_Float16)a2.w;
    A[12] = (_Float16)a3.x; A[13] = (_Float16)a3.y;
    A[14] = (_Float16)a3.z; A[15] = (_Float16)a3.w;
  }

  v8f z0 = {}, z1 = {};
  v16h B0 = *(const v16h*)&wr[(n)      * CW + hi * 16];
  v8f c0 = __builtin_amdgcn_wmma_f32_16x16x32_f16(false, A, false, B0, (short)0, z0, false, false);
  v16h B1 = *(const v16h*)&wr[(16 + n) * CW + hi * 16];
  v8f c1 = __builtin_amdgcn_wmma_f32_16x16x32_f16(false, A, false, B1, (short)0, z1, false, false);

  float wo_lo = W_out[n],      wo_hi = W_out[16 + n];
  float bc_lo = b_conv[n],     bc_hi = b_conv[16 + n];

  float partial[8];
#pragma unroll
  for (int r = 0; r < 8; ++r) {
    int m   = mtile * 16 + hi * 8 + r;
    int mcl = m < NN ? m : NN - 1;
    float lo = lrelu(aggr[(size_t)mcl * CW + n]      + c0[r] + bc_lo);
    float hv = lrelu(aggr[(size_t)mcl * CW + 16 + n] + c1[r] + bc_hi);
    partial[r] = fmaf(lo, wo_lo, hv * wo_hi);
  }
#pragma unroll
  for (int r = 0; r < 8; ++r) {
    float s = partial[r];
    s += __shfl_xor(s, 1);
    s += __shfl_xor(s, 2);
    s += __shfl_xor(s, 4);
    s += __shfl_xor(s, 8);
    partial[r] = s;
  }
  if (n == 0) {
    float bo = b_out[0];
#pragma unroll
    for (int r = 0; r < 8; ++r) {
      int m = mtile * 16 + hi * 8 + r;
      if (m < NN) out[m] = partial[r] + bo;
    }
  }
}

// ---------------------------------------------------------------------------
extern "C" void kernel_launch(void* const* d_in, const int* in_sizes, int n_in,
                              void* d_out, int out_size, void* d_ws, size_t ws_size,
                              hipStream_t stream) {
  const float* x         = (const float*)d_in[0];
  const int*   edge_index= (const int*)  d_in[1];
  const float* edge_attr = (const float*)d_in[2];
  const float* W_in      = (const float*)d_in[3];
  const float* b_in      = (const float*)d_in[4];
  const float* W_edge    = (const float*)d_in[5];
  const float* b_edge    = (const float*)d_in[6];
  const float* W_root    = (const float*)d_in[7];
  const float* b_conv    = (const float*)d_in[8];
  const float* W_out     = (const float*)d_in[9];
  const float* b_out     = (const float*)d_in[10];
  float* out = (float*)d_out;

  float* h    = (float*)d_ws;                    // [NN, CW]
  float* aggr = h + (size_t)NN * CW;             // [NN, CW]

  hipMemsetAsync(aggr, 0, (size_t)NN * CW * sizeof(float), stream);

  const int mtiles  = (NN + 15) / 16;
  const int blocks1 = (mtiles + 7) / 8;

  k_input_proj<<<blocks1, 256, 0, stream>>>(x, W_in, b_in, h);
  k_edge_fused<<<EE / 128, 256, 0, stream>>>(edge_attr, edge_index,
                                             W_edge, b_edge, h, aggr);
  k_root_out<<<blocks1, 256, 0, stream>>>(h, aggr, W_root, b_conv,
                                          W_out, b_out, out);
}
